// CL4KTTransformerLayer_88381837017140
// MI455X (gfx1250) — compile-verified, matrix operand order checked by reading, not run on hardware
//
#include <hip/hip_runtime.h>
#include <hip/hip_bf16.h>
#include <stdint.h>

// ---------------------------------------------------------------------------
// Problem constants (reference: B,S,D,H = 64,512,1024,16; DK=64; DFF=4096)
// ---------------------------------------------------------------------------
#define B_   64
#define S_   512
#define D_   1024
#define H_   16
#define DK_  64
#define DFF_ 4096

typedef __bf16 bf16;
typedef __attribute__((ext_vector_type(16))) __bf16 bf16x16;
typedef __attribute__((ext_vector_type(8)))  __bf16 bf16x8;
typedef __attribute__((ext_vector_type(8)))  float  f32x8;

static __device__ inline f32x8 zero8() {
  f32x8 z;
#pragma unroll
  for (int e = 0; e < 8; ++e) z[e] = 0.0f;
  return z;
}

static __device__ inline f32x8 wmma_bf16(bf16x16 a, bf16x16 b, f32x8 c) {
  // v_wmma_f32_16x16x32_bf16 : D = A(16x32 bf16) * B(32x16 bf16) + C(f32)
  return __builtin_amdgcn_wmma_f32_16x16x32_bf16(
      /*neg_a=*/false, a, /*neg_b=*/false, b,
      /*c_mod=*/(short)0, c, /*reuse_a=*/false, /*reuse_b=*/false);
}

// Async global->LDS copy of 16 bytes per lane (ASYNCcnt-tracked, no VGPR data).
// The LDS destination is the byte address held in the VDST VGPR; low 32 bits of a generic LDS pointer are the
// LDS byte offset (aperture in the high bits), so truncation gives VDST.
static __device__ inline void async_copy_b128(uint32_t lds_off, const void* gaddr) {
  asm volatile("global_load_async_to_lds_b128 %0, %1, off"
               :: "v"(lds_off), "v"((uint64_t)(uintptr_t)gaddr)
               : "memory");
}
static __device__ inline void wait_asynccnt0() {
  asm volatile("s_wait_asynccnt 0" ::: "memory");
}

// A / B fragment loader (CDNA5 16-bit 16x32 layout):
// lane<16 : row = lane,    elems 0-7 = K 0..7,  elems 8-15 = K 16..23
// lane>=16: row = lane-16, elems 0-7 = K 8..15, elems 8-15 = K 24..31
// `base` points at a row-major [16 x >=32] region with leading dim `ld`.
static __device__ inline bf16x16 load_frag_bf16(const bf16* base, int ld, int lane) {
  const int r  = lane & 15;
  const int kb = (lane >> 4) * 8;
  const bf16* p = base + (size_t)r * ld + kb;
  union { bf16x16 v; bf16x8 h[2]; } u;
  u.h[0] = *(const bf16x8*)(p);
  u.h[1] = *(const bf16x8*)(p + 16);
  return u.v;
}

// Same fragment but converting from an fp32 source on the fly.
static __device__ inline bf16x16 load_frag_f32(const float* base, int ld, int lane) {
  const int r  = lane & 15;
  const int kb = (lane >> 4) * 8;
  const float* p = base + (size_t)r * ld + kb;
  bf16x16 v;
#pragma unroll
  for (int j = 0; j < 8; ++j) v[j] = (bf16)p[j];
#pragma unroll
  for (int j = 0; j < 8; ++j) v[8 + j] = (bf16)p[16 + j];
  return v;
}

// ---------------------------------------------------------------------------
// Weight prep: fp32 W[K][N] -> bf16 WT[N][K] (transpose+convert, once per call)
// so GEMM B-tiles are contiguous 16B copies (no scalar LDS transpose).
// ---------------------------------------------------------------------------
__global__ void cvt_transpose_w(const float* __restrict__ src, bf16* __restrict__ dst,
                                int K, int N) {
  size_t i = (size_t)blockIdx.x * blockDim.x + threadIdx.x;
  const size_t total = (size_t)K * N;
  const size_t stride = (size_t)gridDim.x * blockDim.x;
  for (; i < total; i += stride) {
    const int k = (int)(i / N);
    const int n = (int)(i - (size_t)k * N);
    dst[(size_t)n * K + k] = (bf16)src[i];
  }
}

// v (B,H,S,DK) bf16 -> vT (B,H,DK,S) bf16 so ctx GEMM B-fragments are contiguous
__global__ void transpose_v(const bf16* __restrict__ vs, bf16* __restrict__ vT) {
  size_t i = (size_t)blockIdx.x * blockDim.x + threadIdx.x;
  const size_t total = (size_t)B_ * H_ * S_ * DK_;
  if (i >= total) return;
  size_t bh = i >> 15;      // / (S*DK)
  size_t r  = i & 32767;
  size_t s  = r >> 6;       // / DK
  size_t dk = r & 63;
  vT[(bh << 15) + (dk << 9) + s] = vs[i];
}

// ---------------------------------------------------------------------------
// Generic WMMA GEMM:  out = epilogue( A(MxK) @ W(KxN) + bias ),  W given as
// WT (NxK).  Block tile 128x128, K chunk 32, 256 threads = 8 waves (2x4),
// each wave: 64x32 -> 4x2 accumulators of 16x16.
// Double-buffered: tile (it+1) is staged with global_load_async_to_lds_b128
// while tile (it) feeds the WMMA burst; one barrier per K chunk.
// Template: AF32  - A is fp32, convert while staging to LDS
//           SPLIT - bf16 out in (B,H,S,DK) layout
//           GELU  - exact gelu epilogue
//           RESID/OUTF/OUTH - residual add / fp32 out / bf16 out
// ---------------------------------------------------------------------------
template<bool AF32, bool SPLIT, bool GELU, bool RESID, bool OUTF, bool OUTH>
__global__ __launch_bounds__(256)
void gemm_wmma(const void* __restrict__ Aptr, const bf16* __restrict__ WT,
               const float* __restrict__ bias, const float* __restrict__ resid,
               float* __restrict__ outF, bf16* __restrict__ outH,
               int M, int N, int K)
{
  __shared__ __align__(16) bf16 As[2][128 * 32];   // [m][k]
  __shared__ __align__(16) bf16 Bs[2][128 * 32];   // [n][k]

  const int tid  = threadIdx.x;
  const int lane = tid & 31;
  const int wave = tid >> 5;
  const int wm   = wave >> 2;          // 0..1
  const int wn   = wave & 3;           // 0..3
  const int mBlk = blockIdx.y * 128;
  const int nBlk = blockIdx.x * 128;

  f32x8 acc[4][2];
#pragma unroll
  for (int i = 0; i < 4; ++i)
#pragma unroll
    for (int j = 0; j < 2; ++j) acc[i][j] = zero8();

  const bf16*  Ah = (const bf16*)Aptr;
  const float* Af = (const float*)Aptr;

  // each thread owns one 16-element half-row of each tile
  const int r  = tid >> 1;             // 0..127
  const int c0 = (tid & 1) * 16;       // 0 or 16
  const uint32_t ldsA = (uint32_t)(uintptr_t)&As[0][r * 32 + c0];
  const uint32_t ldsB = (uint32_t)(uintptr_t)&Bs[0][r * 32 + c0];
  const uint32_t bufStep = 128 * 32 * 2;   // bytes between buffers

  // stage K-chunk k0 into buffer `buf`
  auto issue = [&](int buf, int k0) {
    if constexpr (AF32) {
      const float* src = Af + (size_t)(mBlk + r) * K + k0 + c0;
#pragma unroll
      for (int j = 0; j < 16; ++j) As[buf][r * 32 + c0 + j] = (bf16)src[j];
    } else {
      const bf16* src = Ah + (size_t)(mBlk + r) * K + k0 + c0;
      async_copy_b128(ldsA + buf * bufStep,      src);
      async_copy_b128(ldsA + buf * bufStep + 16, src + 8);
    }
    const bf16* wsrc = WT + (size_t)(nBlk + r) * K + k0 + c0;
    async_copy_b128(ldsB + buf * bufStep,      wsrc);
    async_copy_b128(ldsB + buf * bufStep + 16, wsrc + 8);
  };

  const int nIter = K >> 5;
  issue(0, 0);
  for (int it = 0; it < nIter; ++it) {
    const int cur = it & 1;
    wait_asynccnt0();        // this wave's copies into buf[cur] are complete
    __syncthreads();         // everyone's copies landed; buf[cur^1] fully read
    if (it + 1 < nIter) issue(cur ^ 1, (it + 1) << 5);

    bf16x16 bfrag[2];
#pragma unroll
    for (int ni = 0; ni < 2; ++ni)
      bfrag[ni] = load_frag_bf16(&Bs[cur][(wn * 32 + ni * 16) * 32], 32, lane);
#pragma unroll
    for (int mi = 0; mi < 4; ++mi) {
      bf16x16 afrag = load_frag_bf16(&As[cur][(wm * 64 + mi * 16) * 32], 32, lane);
#pragma unroll
      for (int ni = 0; ni < 2; ++ni)
        acc[mi][ni] = wmma_bf16(afrag, bfrag[ni], acc[mi][ni]);
    }
  }

  // epilogue (C/D layout: row = vgpr + 8*(lane>=16), col = lane&15)
  const int colLane = lane & 15;
  const int rowOff  = (lane >> 4) * 8;
#pragma unroll
  for (int mi = 0; mi < 4; ++mi) {
#pragma unroll
    for (int ni = 0; ni < 2; ++ni) {
      const int mBase = mBlk + wm * 64 + mi * 16;
      const int nBase = nBlk + wn * 32 + ni * 16;
      const int col = nBase + colLane;
      const float bvv = bias ? bias[col] : 0.0f;
#pragma unroll
      for (int e = 0; e < 8; ++e) {
        const int row = mBase + rowOff + e;
        const size_t pidx = (size_t)row * N + col;
        float v = acc[mi][ni][e] + bvv;
        if constexpr (RESID) v += resid[pidx];
        if constexpr (GELU)  v = 0.5f * v * (1.0f + erff(v * 0.70710678118f));
        if constexpr (OUTF)  outF[pidx] = v;
        if constexpr (OUTH) {
          size_t oidx;
          if constexpr (SPLIT) {
            const int b = row >> 9, s = row & 511;   // S = 512
            const int h = col >> 6, dk = col & 63;   // DK = 64
            oidx = ((((size_t)b * H_ + h) * S_) + s) * DK_ + dk;
          } else {
            oidx = pidx;
          }
          outH[oidx] = (bf16)v;
        }
      }
    }
  }
}

// ---------------------------------------------------------------------------
// scores = q @ k^T / sqrt(DK) -> fp32 into p_attn region (used as scratch)
// grid (B*H, S/16); 8 waves; wave w owns 4 j-tiles (w*64..w*64+63)
// ---------------------------------------------------------------------------
__global__ __launch_bounds__(256)
void attn_qk(const bf16* __restrict__ qs, const bf16* __restrict__ ks,
             float* __restrict__ pat)
{
  const int bh    = blockIdx.x;
  const int iBase = blockIdx.y * 16;
  const int lane  = threadIdx.x & 31;
  const int wave  = threadIdx.x >> 5;
  const int jBase = wave * 64;
  const bf16* qb = qs + (size_t)bh * S_ * DK_;
  const bf16* kb = ks + (size_t)bh * S_ * DK_;

  f32x8 acc[4];
#pragma unroll
  for (int jt = 0; jt < 4; ++jt) acc[jt] = zero8();

#pragma unroll
  for (int kc = 0; kc < DK_; kc += 32) {
    bf16x16 a = load_frag_bf16(qb + (size_t)iBase * DK_ + kc, DK_, lane);
#pragma unroll
    for (int jt = 0; jt < 4; ++jt) {
      bf16x16 b = load_frag_bf16(kb + (size_t)(jBase + jt * 16) * DK_ + kc, DK_, lane);
      acc[jt] = wmma_bf16(a, b, acc[jt]);
    }
  }
  float* pb = pat + (size_t)bh * S_ * S_;
  const int r0 = iBase + (lane >> 4) * 8;
#pragma unroll
  for (int jt = 0; jt < 4; ++jt) {
    const int col = jBase + jt * 16 + (lane & 15);
#pragma unroll
    for (int e = 0; e < 8; ++e)
      pb[(size_t)(r0 + e) * S_ + col] = acc[jt][e] * 0.125f;  // 1/sqrt(64)
  }
}

// ---------------------------------------------------------------------------
// Per-row: masked softmax -> cumsum -> total-effect -> re-masked softmax.
// One 256-thread block per (b,h,i) row of 512; in-place in pat.
// ---------------------------------------------------------------------------
__global__ __launch_bounds__(256)
void attn_te_softmax(float* __restrict__ pat, const float* __restrict__ gammas,
                     const int* __restrict__ maskp)
{
  const int i  = blockIdx.x;
  const int bh = blockIdx.y;
  const int h  = bh & (H_ - 1);
  float* row = pat + (size_t)bh * S_ * S_ + (size_t)i * S_;
  const int t = threadIdx.x;
  const int mask = maskp ? *maskp : 1;

  __shared__ float red[256];

  const int j0 = 2 * t, j1 = 2 * t + 1;
  const float sc0 = row[j0], sc1 = row[j1];
  const bool a0 = (j0 - i) < mask;
  const bool a1 = (j1 - i) < mask;
  const float x0 = a0 ? sc0 : -1e32f;
  const float x1 = a1 ? sc1 : -1e32f;

  // softmax #1 (over masked scores)
  red[t] = fmaxf(x0, x1);
  __syncthreads();
  for (int off = 128; off > 0; off >>= 1) {
    if (t < off) red[t] = fmaxf(red[t], red[t + off]);
    __syncthreads();
  }
  const float rmax = red[0];
  __syncthreads();
  const float e0 = expf(x0 - rmax), e1 = expf(x1 - rmax);
  red[t] = e0 + e1;
  __syncthreads();
  for (int off = 128; off > 0; off >>= 1) {
    if (t < off) red[t] += red[t + off];
    __syncthreads();
  }
  const float Z = red[0];
  __syncthreads();
  const float s0 = (e0 / Z) * (a0 ? 1.0f : 0.0f);
  const float s1 = (e1 / Z) * (a1 ? 1.0f : 0.0f);

  // inclusive scan of per-thread pair sums (Hillis-Steele)
  const float pair = s0 + s1;
  red[t] = pair;
  __syncthreads();
  for (int off = 1; off < 256; off <<= 1) {
    const float v = red[t];
    const float w = (t >= off) ? red[t - off] : 0.0f;
    __syncthreads();
    red[t] = v + w;
    __syncthreads();
  }
  const float incl = red[t];
  const float tot  = red[255];
  __syncthreads();
  const float excl = incl - pair;
  const float dc0 = excl + s0;
  const float dc1 = excl + s0 + s1;

  // gamma = -softplus(gammas[h])
  const float g  = gammas[h];
  const float sp = (g > 0.0f) ? (g + log1pf(expf(-g))) : log1pf(expf(g));
  const float gamma = -sp;

  const float p0f = fabsf((float)(i - j0));
  const float p1f = fabsf((float)(i - j1));
  const float d0 = sqrtf(fmaxf((tot - dc0) * p0f, 0.0f));
  const float d1 = sqrtf(fmaxf((tot - dc1) * p1f, 0.0f));
  const float te0 = fminf(fmaxf(expf(d0 * gamma), 1e-5f), 1e5f);
  const float te1 = fminf(fmaxf(expf(d1 * gamma), 1e-5f), 1e5f);

  const float y0 = a0 ? sc0 * te0 : -1e32f;
  const float y1 = a1 ? sc1 * te1 : -1e32f;

  // softmax #2
  red[t] = fmaxf(y0, y1);
  __syncthreads();
  for (int off = 128; off > 0; off >>= 1) {
    if (t < off) red[t] = fmaxf(red[t], red[t + off]);
    __syncthreads();
  }
  const float m2 = red[0];
  __syncthreads();
  const float f0 = expf(y0 - m2), f1 = expf(y1 - m2);
  red[t] = f0 + f1;
  __syncthreads();
  for (int off = 128; off > 0; off >>= 1) {
    if (t < off) red[t] += red[t + off];
    __syncthreads();
  }
  const float Z2 = red[0];
  row[j0] = f0 / Z2;
  row[j1] = f1 / Z2;
}

// ---------------------------------------------------------------------------
// ctx = p_attn @ v  ->  bf16 concat layout (B,S,D)
// grid (B*H, S/128); wave w owns rows [blockIdx.y*128 + w*16, +16)
// ---------------------------------------------------------------------------
__global__ __launch_bounds__(256)
void attn_ctx(const float* __restrict__ pat, const bf16* __restrict__ vT,
              bf16* __restrict__ ctx)
{
  const int bh = blockIdx.x;
  const int b = bh >> 4, h = bh & 15;
  const int lane = threadIdx.x & 31;
  const int wave = threadIdx.x >> 5;
  const int iBase = blockIdx.y * 128 + wave * 16;
  const float* pb = pat + (size_t)bh * S_ * S_;
  const bf16*  vb = vT + (size_t)bh * DK_ * S_;   // [dk][j]

  f32x8 acc[4];
#pragma unroll
  for (int nt = 0; nt < 4; ++nt) acc[nt] = zero8();

  for (int kc = 0; kc < S_; kc += 32) {
    bf16x16 a = load_frag_f32(pb + (size_t)iBase * S_ + kc, S_, lane);
#pragma unroll
    for (int nt = 0; nt < 4; ++nt) {
      bf16x16 bfr = load_frag_bf16(vb + (size_t)(nt * 16) * S_ + kc, S_, lane);
      acc[nt] = wmma_bf16(a, bfr, acc[nt]);
    }
  }
  const int r0 = iBase + (lane >> 4) * 8;
#pragma unroll
  for (int nt = 0; nt < 4; ++nt) {
    const int dk = nt * 16 + (lane & 15);
#pragma unroll
    for (int e = 0; e < 8; ++e)
      ctx[((size_t)(b * S_ + r0 + e)) * D_ + h * DK_ + dk] = (bf16)acc[nt][e];
  }
}

// ---------------------------------------------------------------------------
// Row layernorm over D=1024: two-pass mean/var, fp32 out (+ optional bf16)
// ---------------------------------------------------------------------------
__global__ __launch_bounds__(256)
void layernorm_rows(const float* __restrict__ x, const float* __restrict__ w,
                    const float* __restrict__ b, float* __restrict__ outF,
                    bf16* __restrict__ outH)
{
  const int rowi = blockIdx.x;
  const float* xr = x + (size_t)rowi * D_;
  const int t = threadIdx.x;
  __shared__ float red[256];

  const float v0 = xr[t], v1 = xr[t + 256], v2 = xr[t + 512], v3 = xr[t + 768];
  red[t] = v0 + v1 + v2 + v3;
  __syncthreads();
  for (int off = 128; off > 0; off >>= 1) {
    if (t < off) red[t] += red[t + off];
    __syncthreads();
  }
  const float mean = red[0] * (1.0f / 1024.0f);
  __syncthreads();
  const float d0 = v0 - mean, d1 = v1 - mean, d2 = v2 - mean, d3 = v3 - mean;
  red[t] = d0 * d0 + d1 * d1 + d2 * d2 + d3 * d3;
  __syncthreads();
  for (int off = 128; off > 0; off >>= 1) {
    if (t < off) red[t] += red[t + off];
    __syncthreads();
  }
  const float var = red[0] * (1.0f / 1024.0f);
  const float inv = rsqrtf(var + 1e-5f);

  const size_t base = (size_t)rowi * D_;
#pragma unroll
  for (int q = 0; q < 4; ++q) {
    const int c = t + q * 256;
    const float dv = (q == 0 ? d0 : q == 1 ? d1 : q == 2 ? d2 : d3);
    const float y = dv * inv * w[c] + b[c];
    if (outF) outF[base + c] = y;
    if (outH) outH[base + c] = (bf16)y;
  }
}

// ---------------------------------------------------------------------------
// Host-side orchestration
// ---------------------------------------------------------------------------
extern "C" void kernel_launch(void* const* d_in, const int* in_sizes, int n_in,
                              void* d_out, int out_size, void* d_ws, size_t ws_size,
                              hipStream_t stream)
{
  (void)in_sizes; (void)n_in; (void)out_size; (void)ws_size;

  const float* query  = (const float*)d_in[0];
  const float* keys   = (const float*)d_in[1];
  const float* values = (const float*)d_in[2];
  const float* Wk = (const float*)d_in[3];
  const float* bk = (const float*)d_in[4];
  const float* Wv = (const float*)d_in[5];
  const float* bv = (const float*)d_in[6];
  const float* Wo = (const float*)d_in[7];
  const float* bo = (const float*)d_in[8];
  const float* gammas = (const float*)d_in[9];
  const float* ln1w = (const float*)d_in[10];
  const float* ln1b = (const float*)d_in[11];
  const float* W1 = (const float*)d_in[12];
  const float* b1 = (const float*)d_in[13];
  const float* W2 = (const float*)d_in[14];
  const float* b2 = (const float*)d_in[15];
  const float* ln2w = (const float*)d_in[16];
  const float* ln2b = (const float*)d_in[17];
  const int*   maskp = (const int*)d_in[18];

  float* out  = (float*)d_out;
  float* xOut = out;                               // (B,S,D)
  float* pat  = out + (size_t)B_ * S_ * D_;        // (B,H,S,S), also scores scratch

  const size_t BSD   = (size_t)B_ * S_ * D_;
  const size_t BSDFF = (size_t)B_ * S_ * DFF_;

  char* wsb = (char*)d_ws;
  size_t off = 0;
  auto alloc = [&](size_t bytes) -> void* {
    void* p = wsb + off;
    off += (bytes + 255) & ~(size_t)255;
    return p;
  };
  bf16* qs    = (bf16*)alloc(BSD * 2);             // (B,H,S,DK)
  bf16* ks    = (bf16*)alloc(BSD * 2);
  bf16* vs    = (bf16*)alloc(BSD * 2);
  bf16* vT    = (bf16*)alloc(BSD * 2);             // (B,H,DK,S)
  bf16* ctx   = (bf16*)alloc(BSD * 2);             // (B,S,D)
  bf16* WkT   = (bf16*)alloc((size_t)D_ * D_ * 2);    // transposed bf16 weights
  bf16* WvT   = (bf16*)alloc((size_t)D_ * D_ * 2);
  bf16* WoT   = (bf16*)alloc((size_t)D_ * D_ * 2);
  bf16* W1T   = (bf16*)alloc((size_t)D_ * DFF_ * 2);  // (DFF x D)
  bf16* W2T   = (bf16*)alloc((size_t)DFF_ * D_ * 2);  // (D x DFF)
  float* x1   = (float*)alloc(BSD * 4);            // query + attn_out (pre-LN1)
  float* xln  = (float*)alloc(BSD * 4);            // post-LN1 fp32
  bf16* xln_h = (bf16*)alloc(BSD * 2);             // post-LN1 bf16
  bf16* h1    = (bf16*)alloc(BSDFF * 2);           // gelu(x@W1+b1) bf16
  float* x2   = (float*)alloc(BSD * 4);            // xln + ff (pre-LN2)

  const int M = B_ * S_;   // 32768
  dim3 blk(256);

  // 1) weights -> transposed bf16
  cvt_transpose_w<<<2048, 256, 0, stream>>>(Wk, WkT, D_, D_);
  cvt_transpose_w<<<2048, 256, 0, stream>>>(Wv, WvT, D_, D_);
  cvt_transpose_w<<<2048, 256, 0, stream>>>(Wo, WoT, D_, D_);
  cvt_transpose_w<<<2048, 256, 0, stream>>>(W1, W1T, D_, DFF_);
  cvt_transpose_w<<<2048, 256, 0, stream>>>(W2, W2T, DFF_, D_);

  // 2) projections (A fp32 converted in-LDS, split-head bf16 outputs)
  gemm_wmma<true, true, false, false, false, true>
      <<<dim3(D_ / 128, M / 128), blk, 0, stream>>>(
      query,  WkT, bk, nullptr, nullptr, qs, M, D_, D_);
  gemm_wmma<true, true, false, false, false, true>
      <<<dim3(D_ / 128, M / 128), blk, 0, stream>>>(
      keys,   WkT, bk, nullptr, nullptr, ks, M, D_, D_);
  gemm_wmma<true, true, false, false, false, true>
      <<<dim3(D_ / 128, M / 128), blk, 0, stream>>>(
      values, WvT, bv, nullptr, nullptr, vs, M, D_, D_);
  transpose_v<<<(unsigned)((BSD + 255) / 256), 256, 0, stream>>>(vs, vT);

  // 3) attention
  attn_qk<<<dim3(B_ * H_, S_ / 16), blk, 0, stream>>>(qs, ks, pat);
  attn_te_softmax<<<dim3(S_, B_ * H_), blk, 0, stream>>>(pat, gammas, maskp);
  attn_ctx<<<dim3(B_ * H_, S_ / 128), blk, 0, stream>>>(pat, vT, ctx);

  // 4) output projection + residual, LN1
  gemm_wmma<false, false, false, true, true, false>
      <<<dim3(D_ / 128, M / 128), blk, 0, stream>>>(
      ctx, WoT, bo, query, x1, nullptr, M, D_, D_);
  layernorm_rows<<<M, blk, 0, stream>>>(x1, ln1w, ln1b, xln, xln_h);

  // 5) FFN + residual, LN2 (writes final x into d_out)
  gemm_wmma<false, false, true, false, false, true>
      <<<dim3(DFF_ / 128, M / 128), blk, 0, stream>>>(
      xln_h, W1T, b1, nullptr, nullptr, h1, M, DFF_, D_);
  gemm_wmma<false, false, false, true, true, false>
      <<<dim3(D_ / 128, M / 128), blk, 0, stream>>>(
      h1, W2T, b2, xln, x2, nullptr, M, D_, DFF_);
  layernorm_rows<<<M, blk, 0, stream>>>(x2, ln2w, ln2b, xOut, nullptr);
}
